// NetVLAD_55619826483530
// MI455X (gfx1250) — matrix operand order repaired and checked by reading, not compiled, not run on hardware
//
#include <hip/hip_runtime.h>
#include <math.h>

// ---------------- CDNA5 WMMA vector types ----------------
typedef __attribute__((ext_vector_type(8)))  float   v8f;
typedef __attribute__((ext_vector_type(16))) __bf16  v16bf;
typedef __attribute__((ext_vector_type(8)))  __bf16  v8bf;

#define CAT16(lo, hi) __builtin_shufflevector(lo, hi, 0,1,2,3,4,5,6,7,8,9,10,11,12,13,14,15)

// ---------------- Problem constants ----------------
#define Bb   64
#define Dd   512
#define Nn   1200          // H*W = 30*40
#define Kk   64
#define NT   96            // pixels per workgroup tile (6 n-tiles of 16, 3 n-steps of 32)
#define NCH  13            // ceil(1200/96)

// LDS row strides (elements). All byte strides are 16B multiples (aligned v8bf
// loads) and odd-dword (52/260 dwords -> 16 lanes hit 16 distinct banks).
#define LDN   104          // xt_dn [Dd][LDN]  bf16 : rows along n
#define LDND  520          // xt_nd [NT][LDND] bf16 : rows along d (transposed copy)
#define LDNF  100          // lg    [Kk][LDNF] f32  : logits / softmax
#define LDNA  104          // abf   [Kk][LDNA] bf16 : assignment weights

#define XT_DN_BYTES  (Dd * LDN * 2)            // 106,496
#define XT_ND_BYTES  (NT * LDND * 2)           //  99,840
#define LG_BYTES     (Kk * LDNF * 4)           //  25,600
#define ABF_BYTES    (Kk * LDNA * 2)           //  13,312
#define RNS_BYTES    (NT * 4)                  //     384
#define SMEM_BYTES   (XT_DN_BYTES + XT_ND_BYTES + LG_BYTES + ABF_BYTES + RNS_BYTES) // 245,632

// ---------------- Workspace layout (floats unless noted) ----------------
#define VLAD_OFF   0
#define ASUM_OFF   (Bb*Kk*Dd)               // 2,097,152
#define BNORM_OFF  (ASUM_OFF + Bb*Kk)       // 2,101,248
#define ZEROF      (BNORM_OFF + Bb)         // 2,101,312 floats zeroed per call
#define WBF_BYTE_OFF ((size_t)ZEROF * 4)    // bf16 conv_w copy after the f32 region

// =====================================================================
// Kernel 0: zero accumulators, convert conv_w (K x D) fp32 -> bf16
// =====================================================================
__global__ void k_init(const float* __restrict__ conv_w,
                       float* __restrict__ wsf, __bf16* __restrict__ wbf) {
  int i = blockIdx.x * 256 + threadIdx.x;
  if (i < ZEROF) wsf[i] = 0.0f;
  if (i < Kk * Dd) wbf[i] = (__bf16)conv_w[i];
}

// =====================================================================
// Kernel 1 (main): per (b, n-chunk): normalize -> GEMM1 -> softmax ->
//                  asum -> GEMM2 -> atomic vlad accumulation
// =====================================================================
__global__ __launch_bounds__(256)
void k_main(const float* __restrict__ x, const __bf16* __restrict__ wbf,
            float* __restrict__ vlad, float* __restrict__ asum) {
  extern __shared__ char smem[];
  __bf16* xt_dn = (__bf16*)smem;                                   // [Dd][LDN]
  __bf16* xt_nd = (__bf16*)(smem + XT_DN_BYTES);                   // [NT][LDND]
  float*  lg    = (float*) (smem + XT_DN_BYTES + XT_ND_BYTES);     // [Kk][LDNF]
  __bf16* abf   = (__bf16*)(smem + XT_DN_BYTES + XT_ND_BYTES + LG_BYTES); // [Kk][LDNA]
  float*  rns   = (float*) (smem + XT_DN_BYTES + XT_ND_BYTES + LG_BYTES + ABF_BYTES);

  const int blk = blockIdx.x;
  const int b   = blk / NCH;
  const int ci  = blk % NCH;
  const int n0  = ci * NT;
  const int tid = threadIdx.x;
  const int wv  = tid >> 5;        // wave id 0..7
  const int ln  = tid & 31;        // lane id
  const int half = ln >> 4;        // lane half (0/1)
  const int lr   = ln & 15;

  const float* xb = x + (size_t)b * Dd * Nn;

  // ---- Phase A: per-pixel reciprocal L2 norm (coalesced across lanes) ----
  if (tid < NT) {
    int gn = n0 + tid;
    if (gn < Nn) {
      const float* px = xb + gn;
      float s = 0.0f;
      #pragma unroll 8
      for (int d = 0; d < Dd; ++d) { float v = px[(size_t)d * Nn]; s += v * v; }
      rns[tid] = 1.0f / fmaxf(sqrtf(s), 1e-12f);
    } else {
      rns[tid] = 0.0f;
    }
  }
  __syncthreads();

  // ---- Phase B: stage normalized tile as bf16 into LDS in BOTH layouts ----
  for (int idx = tid; idx < Dd * NT; idx += 256) {
    int d = idx / NT;
    int n = idx - d * NT;
    int gn = n0 + n;
    float v = 0.0f;
    if (gn < Nn) v = xb[(size_t)d * Nn + gn] * rns[n];   // L2 hit (pass A warmed it)
    __bf16 h = (__bf16)v;
    xt_dn[d * LDN + n] = h;       // n-contiguous rows -> GEMM2 B frags
    xt_nd[n * LDND + d] = h;      // d-contiguous rows -> GEMM1 B frags
  }
  __syncthreads();

  // ---- Phase C: GEMM1  logits[64][96] = W(64x512) * x̂(512x96) ----
  // Wave w owns k-tile (w>>1) and 3 n-tiles ((w&1)*3 ..): the W A-fragment is
  // loaded once per d-chunk and reused for 3 WMMAs.
  {
    const int kt  = wv >> 1;
    const int ntb = (wv & 1) * 3;
    const int m   = kt * 16 + lr;          // weight row (k)
    const int kbA = half * 16;             // A-frag K offset
    const int kbB = half * 16;             // B-frag K offset
    const __bf16* wrow = wbf + (size_t)m * Dd;
    const __bf16* xr0 = xt_nd + ((ntb + 0) * 16 + lr) * LDND;
    const __bf16* xr1 = xt_nd + ((ntb + 1) * 16 + lr) * LDND;
    const __bf16* xr2 = xt_nd + ((ntb + 2) * 16 + lr) * LDND;
    v8f c0 = {}, c1 = {}, c2 = {};
    for (int dc = 0; dc < 16; ++dc) {
      const int off = dc * 32;
      // A: elems j -> K = kbA + j : 16 contiguous bf16
      const __bf16* ap = wrow + off + kbA;
      v16bf a = CAT16(*(const v8bf*)ap, *(const v8bf*)(ap + 8));
      // B: elems j -> K = kbB + j : 16 contiguous bf16 (2x ds_load_b128)
      v16bf b0 = CAT16(*(const v8bf*)(xr0 + off + kbB), *(const v8bf*)(xr0 + off + kbB + 8));
      v16bf b1 = CAT16(*(const v8bf*)(xr1 + off + kbB), *(const v8bf*)(xr1 + off + kbB + 8));
      v16bf b2 = CAT16(*(const v8bf*)(xr2 + off + kbB), *(const v8bf*)(xr2 + off + kbB + 8));
      c0 = __builtin_amdgcn_wmma_f32_16x16x32_bf16(false, a, false, b0, (short)0, c0, false, false);
      c1 = __builtin_amdgcn_wmma_f32_16x16x32_bf16(false, a, false, b1, (short)0, c1, false, false);
      c2 = __builtin_amdgcn_wmma_f32_16x16x32_bf16(false, a, false, b2, (short)0, c2, false, false);
    }
    #pragma unroll
    for (int r = 0; r < 8; ++r) {
      const int krow = (kt * 16 + r + half * 8) * LDNF + lr;
      lg[krow + (ntb + 0) * 16] = c0[r];
      lg[krow + (ntb + 1) * 16] = c1[r];
      lg[krow + (ntb + 2) * 16] = c2[r];
    }
  }
  __syncthreads();

  // ---- Phase D: softmax over K per column; write f32 (for asum) + bf16 ----
  if (tid < NT) {
    int gn = n0 + tid;
    if (gn < Nn) {
      float mx = -INFINITY;
      for (int k = 0; k < Kk; ++k) mx = fmaxf(mx, lg[k * LDNF + tid]);
      float sum = 0.0f;
      for (int k = 0; k < Kk; ++k) {
        float e = __expf(lg[k * LDNF + tid] - mx);
        lg[k * LDNF + tid] = e;
        sum += e;
      }
      float inv = 1.0f / sum;
      for (int k = 0; k < Kk; ++k) {
        float av = lg[k * LDNF + tid] * inv;
        lg[k * LDNF + tid] = av;
        abf[k * LDNA + tid] = (__bf16)av;
      }
    } else {
      for (int k = 0; k < Kk; ++k) {
        lg[k * LDNF + tid] = 0.0f;
        abf[k * LDNA + tid] = (__bf16)0.0f;
      }
    }
  }
  __syncthreads();

  // ---- Phase E: asum[b,k] += sum_n a[k,n] (reads lg only; no barrier needed) ----
  if (tid < Kk) {
    float s = 0.0f;
    for (int n = 0; n < NT; ++n) s += lg[tid * LDNF + n];
    atomicAdd(&asum[b * Kk + tid], s);
  }

  // ---- Phase F: GEMM2  vlad[64][512] += a(64x96) * x̂ᵀ(96x512) ----
  // Wave w owns k-tile (w>>1) and 16 d-tiles ((w&1)*16 ..), processed in
  // 2 groups of 8: one a-fragment feeds 8 WMMAs (DS traffic ~2.25 loads/WMMA).
  {
    const int kt  = wv >> 1;
    const int dtb = (wv & 1) * 16;
    const int m   = kt * 16 + lr;       // a row (k)
    const int kbA = half * 16;          // A-frag n offset
    const int kbB = half * 16;          // B-frag n offset
    const __bf16* arow = abf + m * LDNA;
    for (int g = 0; g < 2; ++g) {
      const __bf16* xbase = xt_dn + ((dtb + g * 8) * 16 + lr) * LDN;
      v8f acc[8];
      #pragma unroll
      for (int i = 0; i < 8; ++i) { v8f z = {}; acc[i] = z; }
      #pragma unroll
      for (int s = 0; s < 3; ++s) {
        const __bf16* ap = arow + s * 32 + kbA;
        v16bf a = CAT16(*(const v8bf*)ap, *(const v8bf*)(ap + 8));
        #pragma unroll
        for (int i = 0; i < 8; ++i) {
          const __bf16* bp = xbase + i * (16 * LDN) + s * 32 + kbB;
          v16bf bb = CAT16(*(const v8bf*)bp, *(const v8bf*)(bp + 8));
          acc[i] = __builtin_amdgcn_wmma_f32_16x16x32_bf16(false, a, false, bb,
                                                           (short)0, acc[i], false, false);
        }
      }
      #pragma unroll
      for (int i = 0; i < 8; ++i) {
        const int dcol = (dtb + g * 8 + i) * 16 + lr;
        #pragma unroll
        for (int r = 0; r < 8; ++r) {
          int k = kt * 16 + r + half * 8;
          atomicAdd(&vlad[((size_t)b * Kk + k) * Dd + dcol], acc[i][r]);
        }
      }
    }
  }
}

// =====================================================================
// Kernel 2: per (b,k): subtract asum*centroid, intra-normalize, write out,
//           accumulate batch norm contribution
// =====================================================================
__global__ void k_rowfinish(const float* __restrict__ vlad,
                            const float* __restrict__ asum,
                            const float* __restrict__ cent,
                            float* __restrict__ bnorm,
                            float* __restrict__ out) {
  __shared__ float red[256];
  __shared__ float s_rn;
  int bk = blockIdx.x;            // b*64 + k
  int b = bk >> 6, k = bk & 63;
  int tid = threadIdx.x;
  float av = asum[bk];
  float r0 = vlad[(size_t)bk * Dd + tid]       - av * cent[k * Dd + tid];
  float r1 = vlad[(size_t)bk * Dd + tid + 256] - av * cent[k * Dd + tid + 256];
  red[tid] = r0 * r0 + r1 * r1;
  __syncthreads();
  for (int s = 128; s > 0; s >>= 1) {
    if (tid < s) red[tid] += red[tid + s];
    __syncthreads();
  }
  if (tid == 0) {
    float ss = red[0];
    float rn = 1.0f / fmaxf(sqrtf(ss), 1e-12f);
    s_rn = rn;
    atomicAdd(&bnorm[b], ss * rn * rn);
  }
  __syncthreads();
  float rn = s_rn;
  out[(size_t)bk * Dd + tid]       = r0 * rn;
  out[(size_t)bk * Dd + tid + 256] = r1 * rn;
}

// =====================================================================
// Kernel 3: global L2 normalization per batch
// =====================================================================
__global__ void k_gnorm(const float* __restrict__ bnorm, float* __restrict__ out) {
  int b = blockIdx.x;
  float sc = 1.0f / fmaxf(sqrtf(bnorm[b]), 1e-12f);
  for (int i = threadIdx.x; i < Kk * Dd; i += 256)
    out[(size_t)b * Kk * Dd + i] *= sc;
}

// =====================================================================
extern "C" void kernel_launch(void* const* d_in, const int* in_sizes, int n_in,
                              void* d_out, int out_size, void* d_ws, size_t ws_size,
                              hipStream_t stream) {
  (void)in_sizes; (void)n_in; (void)out_size; (void)ws_size;
  const float* x      = (const float*)d_in[0];   // (B, D, H, W) f32
  const float* conv_w = (const float*)d_in[1];   // (K, D) f32
  const float* cent   = (const float*)d_in[2];   // (K, D) f32
  float* out = (float*)d_out;                    // (B, K*D) f32

  float*   wsf   = (float*)d_ws;
  float*   vlad  = wsf + VLAD_OFF;
  float*   asum  = wsf + ASUM_OFF;
  float*   bnorm = wsf + BNORM_OFF;
  __bf16*  wbf   = (__bf16*)((char*)d_ws + WBF_BYTE_OFF);

  // zero accumulators + bf16 weight conversion
  k_init<<<(ZEROF + 255) / 256, 256, 0, stream>>>(conv_w, wsf, wbf);

  // main fused kernel: 245,632 B dynamic LDS (WGP has 320KB; needs opt-in)
  (void)hipFuncSetAttribute((const void*)k_main,
                            hipFuncAttributeMaxDynamicSharedMemorySize, SMEM_BYTES);
  k_main<<<Bb * NCH, 256, SMEM_BYTES, stream>>>(x, wbf, vlad, asum);

  // finalize
  k_rowfinish<<<Bb * Kk, 256, 0, stream>>>(vlad, asum, cent, bnorm, out);
  k_gnorm<<<Bb, 256, 0, stream>>>(bnorm, out);
}
